// EuclideanCodebook_79018808312070
// MI455X (gfx1250) — matrix operand order repaired and checked by reading, not compiled, not run on hardware
//
#include <hip/hip_runtime.h>

// ---------------------------------------------------------------------------
// EuclideanCodebook quantize for MI455X (gfx1250, wave32, WMMA)
//   scores S = X @ E^T  via bf16 hi/lo split (3x wmma_f32_16x16x32_bf16)
//   argmax_k (2*S - ||e_k||^2), then gather embed[idx]
//   Codebook pre-split to bf16 hi/lo in ws; double-buffered LDS staging with
//   GLOBAL_LOAD_ASYNC_TO_LDS_B128 (ASYNCcnt) when available.
// ---------------------------------------------------------------------------

typedef __attribute__((ext_vector_type(16))) __bf16 v16bf;
typedef __attribute__((ext_vector_type(8)))  float  v8f;
typedef __attribute__((ext_vector_type(4)))  float  v4f;
typedef __attribute__((ext_vector_type(4)))  int    i32x4;

#define DIM    256
#define KCODES 2048
#define NTILES (KCODES / 16)
#define TILE_ELEMS (2 * 16 * DIM)          // hi + lo, 8192 bf16 = 16 KB
#define TILE_BYTES (TILE_ELEMS * 2)

#if defined(__has_builtin)
#  if __has_builtin(__builtin_amdgcn_global_load_async_to_lds_b128)
#    define HAVE_ASYNC_LDS 1
#  endif
#endif
#ifndef HAVE_ASYNC_LDS
#  define HAVE_ASYNC_LDS 0
#endif

// ---- ||e_k||^2 precompute --------------------------------------------------
__global__ void vq_esq_kernel(const float* __restrict__ embed,
                              float* __restrict__ esq) {
    int row = blockIdx.x * blockDim.x + threadIdx.x;
    if (row >= KCODES) return;
    const v4f* p = (const v4f*)(embed + (size_t)row * DIM);
    float s = 0.f;
#pragma unroll 8
    for (int i = 0; i < DIM / 4; ++i) {
        v4f v = p[i];
        s += v.x * v.x + v.y * v.y + v.z * v.z + v.w * v.w;
    }
    esq[row] = s;
}

// ---- codebook fp32 -> bf16 hi/lo, tile-contiguous layout -------------------
// cb layout: [NTILES][ hi[16][DIM] , lo[16][DIM] ]  (16 KB per tile)
__global__ void vq_prep_cb_kernel(const float* __restrict__ embed,
                                  __bf16* __restrict__ cb) {
    int t   = blockIdx.x * blockDim.x + threadIdx.x;   // one per (row, 16-chunk)
    int row = t >> 4;
    int c   = (t & 15) * 16;
    if (row >= KCODES) return;
    const int kt = row >> 4, r = row & 15;
    const float* src = embed + (size_t)row * DIM + c;
    __bf16* hi = cb + (size_t)kt * TILE_ELEMS + r * DIM + c;
    __bf16* lo = hi + 16 * DIM;
    float f[16];
    *(v4f*)&f[0]  = *(const v4f*)(src);
    *(v4f*)&f[4]  = *(const v4f*)(src + 4);
    *(v4f*)&f[8]  = *(const v4f*)(src + 8);
    *(v4f*)&f[12] = *(const v4f*)(src + 12);
#pragma unroll
    for (int e = 0; e < 16; ++e) {
        __bf16 h = (__bf16)f[e];
        hi[e] = h;
        lo[e] = (__bf16)(f[e] - (float)h);
    }
}

#if HAVE_ASYNC_LDS
typedef __attribute__((address_space(1))) i32x4 gi32x4;   // global int4
typedef __attribute__((address_space(3))) i32x4 li32x4;   // LDS int4
__device__ __forceinline__ void vq_async_b128(const void* g, void* l) {
    __builtin_amdgcn_global_load_async_to_lds_b128(
        (gi32x4*)(i32x4*)const_cast<void*>(g),
        (li32x4*)(i32x4*)l,
        0, 0);
}
__device__ __forceinline__ void vq_wait_async_le4() {
    asm volatile("s_wait_asynccnt 0x4" ::: "memory");
}
__device__ __forceinline__ void vq_wait_async_0() {
    asm volatile("s_wait_asynccnt 0x0" ::: "memory");
}
#endif

// ---- main kernel: 8 waves/WG, each wave owns a 16-row M-tile ---------------
__launch_bounds__(256)
__global__ void vq_argmax_kernel(const float* __restrict__ x,
                                 const __bf16* __restrict__ cb,
                                 const float* __restrict__ esq,
                                 int*   __restrict__ out_idx,
                                 float* __restrict__ out_idx_f) {
#if HAVE_ASYNC_LDS
    __shared__ __attribute__((aligned(32))) __bf16 sB[2][TILE_ELEMS]; // 32 KB
#else
    __shared__ __attribute__((aligned(32))) __bf16 sB[1][TILE_ELEMS]; // 16 KB
#endif

    const int tid  = threadIdx.x;
    const int wave = tid >> 5;
    const int lane = tid & 31;
    const int lmod = lane & 15;      // M-row selector (A) / N-col (B,C)
    const int g    = lane >> 4;      // lane half (K-chunk selector)

    const int rowBase = blockIdx.x * 128 + wave * 16;
    const float* xrow = x + (size_t)(rowBase + lmod) * DIM;

    // --- A operand (16 rows of x) resident in VGPRs, bf16 hi/lo split ------
    // A k order chosen here (B matches): lanes 0-15 hold K = 0..15, lanes 16-31
    // hold K = 16..31.
    v16bf a_hi[8], a_lo[8];
#pragma unroll
    for (int db = 0; db < 8; ++db) {
        const int o0 = db * 32 + g * 16;
        const int o1 = db * 32 + g * 16 + 8;
        float f[16];
        *(v4f*)&f[0]  = *(const v4f*)(xrow + o0);
        *(v4f*)&f[4]  = *(const v4f*)(xrow + o0 + 4);
        *(v4f*)&f[8]  = *(const v4f*)(xrow + o1);
        *(v4f*)&f[12] = *(const v4f*)(xrow + o1 + 4);
#pragma unroll
        for (int e = 0; e < 16; ++e) {
            __bf16 h = (__bf16)f[e];
            a_hi[db][e] = h;
            a_lo[db][e] = (__bf16)(f[e] - (float)h);
        }
    }

    float best[8];
    int   bidx[8];
#pragma unroll
    for (int i = 0; i < 8; ++i) { best[i] = -3.402823466e38f; bidx[i] = 0; }

    // Each thread stages a 64 B slice of the 16 KB tile.
    const char* gtile0 = (const char*)cb + (size_t)tid * 64;

#if HAVE_ASYNC_LDS
    // prologue: async copy of tile 0 into buffer 0
#pragma unroll
    for (int j = 0; j < 4; ++j)
        vq_async_b128(gtile0 + j * 16, (char*)&sB[0][0] + tid * 64 + j * 16);
#else
    uint4 r4[4];
#pragma unroll
    for (int j = 0; j < 4; ++j)
        r4[j] = *(const uint4*)(gtile0 + j * 16);
#endif

    for (int kt = 0; kt < NTILES; ++kt) {
#if HAVE_ASYNC_LDS
        const int cur = kt & 1;
        __syncthreads();   // all waves done reading buf[cur^1] (iter kt-1)
        if (kt + 1 < NTILES) {
            const char* gnext = gtile0 + (size_t)(kt + 1) * TILE_BYTES;
#pragma unroll
            for (int j = 0; j < 4; ++j)
                vq_async_b128(gnext + j * 16,
                              (char*)&sB[cur ^ 1][0] + tid * 64 + j * 16);
            vq_wait_async_le4();   // tile kt landed; tile kt+1 still in flight
        } else {
            vq_wait_async_0();
        }
        __syncthreads();   // tile kt visible to all waves
        const __bf16* Bb = &sB[cur][0];
#else
        __syncthreads();   // previous tile fully consumed
        {
            char* l = (char*)&sB[0][0] + tid * 64;
#pragma unroll
            for (int j = 0; j < 4; ++j)
                *(uint4*)(l + j * 16) = r4[j];
        }
        uint4 n4[4] = {r4[0], r4[1], r4[2], r4[3]};
        if (kt + 1 < NTILES) {
            const char* gnext = gtile0 + (size_t)(kt + 1) * TILE_BYTES;
#pragma unroll
            for (int j = 0; j < 4; ++j)
                n4[j] = *(const uint4*)(gnext + j * 16);   // overlaps compute
        }
        __syncthreads();   // includes DScnt wait: tile kt visible
        const __bf16* Bb = &sB[0][0];
#endif

        // --- 16x16 score tile: 8 depth chunks x 3 split-precision WMMAs ----
        v8f acc = {};
#pragma unroll
        for (int db = 0; db < 8; ++db) {
            // B k order chosen here (matches A): lane half g supplies K = g*16 .. g*16+15
            const int fb = db * 32 + g * 16;
            v16bf bhi = *(const v16bf*)(Bb + lmod * DIM + fb);
            v16bf blo = *(const v16bf*)(Bb + 16 * DIM + lmod * DIM + fb);
            acc = __builtin_amdgcn_wmma_f32_16x16x32_bf16(
                      false, a_hi[db], false, bhi, (short)0, acc, false, false);
            acc = __builtin_amdgcn_wmma_f32_16x16x32_bf16(
                      false, a_lo[db], false, bhi, (short)0, acc, false, false);
            acc = __builtin_amdgcn_wmma_f32_16x16x32_bf16(
                      false, a_hi[db], false, blo, (short)0, acc, false, false);
        }

        // C layout: N = lane%16 for all 8 acc regs -> one esq per lane/tile
        const int   code = kt * 16 + lmod;
        const float e2   = esq[code];
#pragma unroll
        for (int i = 0; i < 8; ++i) {
            float s = 2.0f * acc[i] - e2;
            if (s > best[i]) { best[i] = s; bidx[i] = code; }
        }

#if !HAVE_ASYNC_LDS
#pragma unroll
        for (int j = 0; j < 4; ++j) r4[j] = n4[j];
#endif
    }

    // --- argmax reduce across the 16 N-lanes of each half-wave -------------
#pragma unroll
    for (int off = 8; off >= 1; off >>= 1) {
#pragma unroll
        for (int i = 0; i < 8; ++i) {
            float ov = __shfl_xor(best[i], off, 16);
            int   oi = __shfl_xor(bidx[i], off, 16);
            if (ov > best[i] || (ov == best[i] && oi < bidx[i])) {
                best[i] = ov; bidx[i] = oi;
            }
        }
    }

    if (lmod == 0) {
#pragma unroll
        for (int i = 0; i < 8; ++i) {
            int row = rowBase + i + 8 * g;   // acc reg i holds row M = i+8g
            out_idx[row]   = bidx[i];
            out_idx_f[row] = (float)bidx[i];
        }
    }
}

// ---- gather: quantize[row] = embed[idx[row]] -------------------------------
__global__ void vq_gather_kernel(const float* __restrict__ embed,
                                 const int* __restrict__ idx,
                                 float* __restrict__ out) {
    const int r = blockIdx.x * 4 + (threadIdx.x >> 6);
    const int c = (threadIdx.x & 63) * 4;
    const int code = idx[r];
    *(v4f*)(out + (size_t)r * DIM + c) =
        *(const v4f*)(embed + (size_t)code * DIM + c);
}

extern "C" void kernel_launch(void* const* d_in, const int* in_sizes, int n_in,
                              void* d_out, int out_size, void* d_ws, size_t ws_size,
                              hipStream_t stream) {
    const float* x     = (const float*)d_in[0];   // [16,4096,256] fp32
    const float* embed = (const float*)d_in[1];   // [2048,256]   fp32
    const int nrows = in_sizes[0] / DIM;          // 65536

    float* quant = (float*)d_out;                        // [nrows,256]
    float* idx_f = (float*)d_out + (size_t)nrows * DIM;  // [nrows] idx as float

    // workspace: cb bf16 hi/lo (2 MB) | esq (8 KB) | idx (int, 256 KB)
    __bf16* cb  = (__bf16*)d_ws;
    float*  esq = (float*)((char*)d_ws + (size_t)NTILES * TILE_BYTES);
    int*    idx_i = (int*)((char*)esq + KCODES * sizeof(float));

    vq_esq_kernel<<<KCODES / 256, 256, 0, stream>>>(embed, esq);
    vq_prep_cb_kernel<<<(KCODES * 16) / 256, 256, 0, stream>>>(embed, cb);
    vq_argmax_kernel<<<nrows / 128, 256, 0, stream>>>(x, cb, esq, idx_i, idx_f);
    vq_gather_kernel<<<nrows / 4, 256, 0, stream>>>(embed, idx_i, quant);
}
